// GNN_5042291605779
// MI455X (gfx1250) — matrix operand order repaired and checked by reading, not compiled, not run on hardware
//
#include <hip/hip_runtime.h>
#include <hip/hip_bf16.h>
#include <hip/hip_fp16.h>

typedef __attribute__((ext_vector_type(16))) _Float16 v16h;
typedef __attribute__((ext_vector_type(8)))  float    v8f;

#define N_NODES   512
#define DDIM      64
#define BATCH     16
#define NEG_SLOPE 0.2f

// Convert 8 contiguous floats (global or LDS) into elements [off..off+7] of a v16h.
__device__ __forceinline__ void cvt8v(v16h& a, int off, const float* p) {
    float4 u = *(const float4*)(p);
    float4 v = *(const float4*)(p + 4);
    a[off + 0] = (_Float16)u.x; a[off + 1] = (_Float16)u.y;
    a[off + 2] = (_Float16)u.z; a[off + 3] = (_Float16)u.w;
    a[off + 4] = (_Float16)v.x; a[off + 5] = (_Float16)v.y;
    a[off + 6] = (_Float16)v.z; a[off + 7] = (_Float16)v.w;
}

// ---------------------------------------------------------------------------
// Kernel A: xl = x @ Wl.T + bl ; xr = x @ Wr.T + br  for all B*N = 8192 rows.
// One wave computes a 16-row x 64-col tile of BOTH xl and xr.
// Also emits xlhT: per-sample TRANSPOSED f16 copy of xl, laid out [b][d][j]
// so kernel B's WMMA B-matrix loads are contiguous along K (=j).
// 4 waves / block, 128 blocks -> 512 row tiles.
// ---------------------------------------------------------------------------
__global__ __launch_bounds__(128) void gat_transform_kernel(
    const float* __restrict__ x,
    const float* __restrict__ Wl, const float* __restrict__ bl,
    const float* __restrict__ Wr, const float* __restrict__ br,
    float* __restrict__ xl, float* __restrict__ xr,
    _Float16* __restrict__ xlhT)
{
    const int lane  = threadIdx.x & 31;
    const int wave  = threadIdx.x >> 5;
    const int tile  = blockIdx.x * 4 + wave;   // 0..511
    const int m0    = tile * 16;
    const int mrow  = lane & 15;
    const int hi    = lane >> 4;               // 0 or 1
    const int abase = hi * 8;                  // unused: A also uses bbase (contiguous k)
    const int bbase = hi * 16;                 // B-matrix K sub-base

    const int bb = m0 >> 9;                    // sample (tile never crosses samples)
    const int j0 = m0 & (N_NODES - 1);         // node index of first row in sample

    v8f cl[4] = {};
    v8f cr[4] = {};

    const float* xrow = x + (size_t)(m0 + mrow) * DDIM;

    #pragma unroll
    for (int kc = 0; kc < DDIM; kc += 32) {
        v16h a;
        cvt8v(a, 0, xrow + kc + bbase);
        cvt8v(a, 8, xrow + kc + bbase + 8);
        #pragma unroll
        for (int nt = 0; nt < 4; nt++) {
            const int n = nt * 16 + mrow;
            v16h bL, bR;
            cvt8v(bL, 0, Wl + (size_t)n * DDIM + kc + bbase);
            cvt8v(bL, 8, Wl + (size_t)n * DDIM + kc + bbase + 8);
            cvt8v(bR, 0, Wr + (size_t)n * DDIM + kc + bbase);
            cvt8v(bR, 8, Wr + (size_t)n * DDIM + kc + bbase + 8);
            cl[nt] = __builtin_amdgcn_wmma_f32_16x16x32_f16(
                false, a, false, bL, (short)0, cl[nt], false, false);
            cr[nt] = __builtin_amdgcn_wmma_f32_16x16x32_f16(
                false, a, false, bR, (short)0, cr[nt], false, false);
        }
    }

    #pragma unroll
    for (int nt = 0; nt < 4; nt++) {
        const int n = nt * 16 + mrow;
        const float blv = bl[n];
        const float brv = br[n];
        #pragma unroll
        for (int r = 0; r < 8; r++) {
            const int m = m0 + r + hi * 8;           // global row
            const int j = j0 + r + hi * 8;           // node within sample
            const float vl = cl[nt][r] + blv;
            const float vr = cr[nt][r] + brv;
            xl[(size_t)m * DDIM + n] = vl;
            xr[(size_t)m * DDIM + n] = vr;
            // transposed f16 copy: [sample][d][j]
            xlhT[((size_t)(bb * DDIM + n)) * N_NODES + j] = (_Float16)vl;
        }
    }
}

// ---------------------------------------------------------------------------
// Kernel B: per (sample b, 16-row target tile i0):
//   scores e[i,j] = att . lrelu(xr[i]+xl[j])  -> LDS [16][512]
//   softmax over j (max + exp/sum passes)
//   out = (p @ xl_f16) / rowsum + bias   (4 waves x 16-col slabs, WMMA)
//   fused[i] = out[i,:] . Wf + bf
// grid = B * 32 = 512 blocks, 256 threads (8 waves).
// ---------------------------------------------------------------------------
__global__ __launch_bounds__(256) void gat_attention_kernel(
    const float* __restrict__ xl, const float* __restrict__ xr,
    const _Float16* __restrict__ xlhT,
    const float* __restrict__ att, const float* __restrict__ bias,
    const float* __restrict__ Wf, const float* __restrict__ bf,
    float* __restrict__ out)
{
    __shared__ float e_s[16][512];     // scores, then exp(e - max)
    __shared__ float xr_s[16][64];     // xr tile; reused as out tile
    __shared__ float att_s[64];
    __shared__ float red_s[16][16];
    __shared__ float rowmax_s[16];
    __shared__ float rowsum_s[16];

    const int t     = threadIdx.x;
    const int b     = blockIdx.x >> 5;     // sample
    const int itile = blockIdx.x & 31;
    const int i0    = itile * 16;

    // Stage xr tile + att into LDS.
    for (int idx = t; idx < 16 * 64; idx += 256) {
        const int i = idx >> 6, d = idx & 63;
        xr_s[i][d] = xr[(size_t)(b * N_NODES + i0 + i) * DDIM + d];
    }
    if (t < 64) att_s[t] = att[t];

    // Prefetch this thread's xl rows while staging completes.
    const float* xl0 = xl + ((size_t)(b * N_NODES) + t) * DDIM;
    const float* xl1 = xl0 + (size_t)256 * DDIM;
    __builtin_prefetch(xl0, 0, 0);
    __builtin_prefetch(xl1, 0, 0);
    __syncthreads();

    // ---- Phase 1: scores. Thread t owns sources j = {t, t+256}; xl read once.
    {
        float acc0[16], acc1[16];
        #pragma unroll
        for (int i = 0; i < 16; i++) { acc0[i] = 0.f; acc1[i] = 0.f; }

        for (int dc = 0; dc < DDIM; dc += 16) {
            float x0[16], x1[16];
            #pragma unroll
            for (int q = 0; q < 4; q++) {
                *(float4*)&x0[q * 4] = *(const float4*)(xl0 + dc + q * 4);
                *(float4*)&x1[q * 4] = *(const float4*)(xl1 + dc + q * 4);
            }
            #pragma unroll
            for (int i = 0; i < 16; i++) {
                float a0 = acc0[i], a1 = acc1[i];
                #pragma unroll
                for (int dd = 0; dd < 16; dd++) {
                    const float ad = att_s[dc + dd];
                    const float xv = xr_s[i][dc + dd];
                    float v0 = xv + x0[dd]; v0 = (v0 > 0.f) ? v0 : NEG_SLOPE * v0;
                    float v1 = xv + x1[dd]; v1 = (v1 > 0.f) ? v1 : NEG_SLOPE * v1;
                    a0 = fmaf(ad, v0, a0);
                    a1 = fmaf(ad, v1, a1);
                }
                acc0[i] = a0; acc1[i] = a1;
            }
        }
        #pragma unroll
        for (int i = 0; i < 16; i++) {
            e_s[i][t]       = acc0[i];
            e_s[i][t + 256] = acc1[i];
        }
    }
    __syncthreads();

    // ---- Phase 2: softmax stats (16 threads per row).
    const int r = t >> 4, c = t & 15;
    {
        float mx = -3.402823e38f;
        for (int j = c; j < N_NODES; j += 16) mx = fmaxf(mx, e_s[r][j]);
        red_s[r][c] = mx;
    }
    __syncthreads();
    if (c == 0) {
        float mx = red_s[r][0];
        #pragma unroll
        for (int k = 1; k < 16; k++) mx = fmaxf(mx, red_s[r][k]);
        rowmax_s[r] = mx;
    }
    __syncthreads();
    {
        const float mx = rowmax_s[r];
        float sum = 0.f;
        for (int j = c; j < N_NODES; j += 16) {
            const float p = __expf(e_s[r][j] - mx);
            e_s[r][j] = p;
            sum += p;
        }
        red_s[r][c] = sum;
    }
    __syncthreads();
    if (c == 0) {
        float s = 0.f;
        #pragma unroll
        for (int k = 0; k < 16; k++) s += red_s[r][k];
        rowsum_s[r] = s;
    }
    __syncthreads();

    // ---- Phase 3: aggregation GEMM. Waves 0..3 each own a 16-column slab.
    // B-matrix rows come from the transposed f16 xl: 16 contiguous halves per
    // lane per k-chunk (one 32B vector load) instead of 16 strided d16 gathers.
    const int wave = t >> 5, lane = t & 31;
    if (wave < 4) {
        const int mrow  = lane & 15;
        const int hi    = lane >> 4;
        const int abase = hi * 8;
        const int bbase = hi * 16;
        const int n     = wave * 16 + mrow;

        v8f cacc = {};
        const _Float16* bptr = xlhT + ((size_t)(b * DDIM + n)) * N_NODES;

        #pragma unroll 4
        for (int kc = 0; kc < N_NODES; kc += 32) {
            v16h a;
            cvt8v(a, 0, &e_s[mrow][kc + bbase]);
            cvt8v(a, 8, &e_s[mrow][kc + bbase + 8]);
            v16h bm = *(const v16h*)(bptr + kc + bbase);   // 32B aligned
            cacc = __builtin_amdgcn_wmma_f32_16x16x32_f16(
                false, a, false, bm, (short)0, cacc, false, false);
        }

        const float bv = bias[n];
        #pragma unroll
        for (int rr = 0; rr < 8; rr++) {
            const int m = rr + hi * 8;
            xr_s[m][n] = cacc[rr] / rowsum_s[m] + bv;   // reuse xr_s as out tile
        }
    }
    __syncthreads();

    // ---- Phase 4: fusion linear.
    if (t < 16) {
        float acc = bf[0];
        #pragma unroll
        for (int d = 0; d < DDIM; d++) acc = fmaf(xr_s[t][d], Wf[d], acc);
        out[(size_t)b * N_NODES + i0 + t] = acc;
    }
}

extern "C" void kernel_launch(void* const* d_in, const int* in_sizes, int n_in,
                              void* d_out, int out_size, void* d_ws, size_t ws_size,
                              hipStream_t stream) {
    const float* x    = (const float*)d_in[0];
    const float* Wl   = (const float*)d_in[1];
    const float* bl   = (const float*)d_in[2];
    const float* Wr   = (const float*)d_in[3];
    const float* br   = (const float*)d_in[4];
    const float* att  = (const float*)d_in[5];
    const float* bias = (const float*)d_in[6];
    const float* Wf   = (const float*)d_in[7];
    const float* bf   = (const float*)d_in[8];
    float* out        = (float*)d_out;

    const size_t rows = (size_t)BATCH * N_NODES * DDIM;   // 524288 elems
    float*     xl   = (float*)d_ws;
    float*     xr   = xl + rows;
    _Float16*  xlhT = (_Float16*)(xr + rows);

    // Kernel A: 512 row tiles, 4 waves/block.
    gat_transform_kernel<<<128, 128, 0, stream>>>(x, Wl, bl, Wr, br, xl, xr, xlhT);
    // Kernel B: one block per (sample, 16-target-row tile).
    gat_attention_kernel<<<BATCH * 32, 256, 0, stream>>>(xl, xr, xlhT, att, bias,
                                                         Wf, bf, out);
}
